// Layer_Hist_81965155877403
// MI455X (gfx1250) — compile-verified, matrix operand order checked by reading, not run on hardware
//
#include <hip/hip_runtime.h>

// CDNA5 / gfx1250 vector types for WMMA
typedef __attribute__((ext_vector_type(16))) _Float16 v16h;
typedef __attribute__((ext_vector_type(8)))  float    v8f;

#define NBINS  30
#define NBPAD  32        // pad to 32 so the second WMMA M-tile is well-defined (bins 30,31 stay 0)
#define NWAVES 8         // 256 threads = 8 wave32
// BINWIDTH = (6-(-6))/(30-2) = 3/7 ; 1/W = 7/3 ; -LO/W = 6*7/3 = 14 (exact in f32)
#define INVW   (2.3333333333333335f)
#define FMAC   (14.0f)

__global__ __launch_bounds__(256) void hist_rows_kernel(
    const float* __restrict__ x, float* __restrict__ out, int cols)
{
    // Per-wave, per-lane privatized counters:
    //   hist[wave][bin][lane] -> LDS bank = (bin*32+lane) & 63 = lane + 32*(bin&1)
    //   => all 32 lanes of a wave always hit distinct banks: conflict-free ds_add_u32.
    __shared__ unsigned hist[NWAVES][NBPAD][32];
    __shared__ float    sums[NBPAD];

    const int tid  = threadIdx.x;
    const int lane = tid & 31;
    const int wave = tid >> 5;
    const int row  = blockIdx.x;

    // Zero the counters (8*32*32 = 8192 words, 32 per thread).
    unsigned* hflat = &hist[0][0][0];
    #pragma unroll
    for (int i = 0; i < (NWAVES * NBPAD * 32) / 256; ++i)
        hflat[tid + 256 * i] = 0u;
    __syncthreads();

    // -------- main streaming loop: global_load_b128, bin, ds_add_u32 --------
    const float4* xr = (const float4*)(x + (size_t)row * (size_t)cols);
    const int n4 = cols >> 2;

    for (int i = tid; i < n4; i += 256) {
        float4 v = xr[i];
        float vv[4] = { v.x, v.y, v.z, v.w };
        #pragma unroll
        for (int e = 0; e < 4; ++e) {
            // k = clip(floor((x-LO)/W), -1, 28) + 1  in [0, 29]
            float t = fmaf(vv[e], INVW, FMAC);              // v_fmaak
            float m = fminf(fmaxf(floorf(t), -1.0f), 28.0f);
            int   k = (int)(m + 1.0f);                      // exact: m integral
            atomicAdd(&hist[wave][k][lane], 1u);            // ds_add_u32 (no rtn)
        }
    }
    __syncthreads();

    // -------- WMMA reduction: sum[bin] = sum over 8 waves x 32 lanes --------
    // hist viewed as [30 x 256] f16 matrix times ones vector; counts <= 64 so
    // f16 operands are exact, f32 accumulation is exact.
    if (tid < 32) {                      // wave 0 only -> EXEC all ones for WMMA
        const int M  = lane & 15;
        const int hi = lane >> 4;

        v16h ones;
        #pragma unroll
        for (int j = 0; j < 16; ++j) ones[j] = (_Float16)1.0f;

        #pragma unroll
        for (int t = 0; t < 2; ++t) {          // M-tiles: bins 0..15, 16..31
            v8f acc = {};
            #pragma unroll
            for (int c = 0; c < NWAVES; ++c) { // K-chunks: one wave (32 counters)
                // 16-bit A-matrix 16x32 layout (ISA 7.12.2):
                // lanes 0-15: VGPR j<4 -> K=2j,2j+1 ; j>=4 -> K=8+2j,8+2j+1
                // lanes 16-31: same +8
                v16h a;
                #pragma unroll
                for (int j = 0; j < 8; ++j) {
                    int K0 = 2 * j + ((j >= 4) ? 8 : 0) + hi * 8;
                    unsigned c0 = hist[c][t * 16 + M][K0];
                    unsigned c1 = hist[c][t * 16 + M][K0 + 1];
                    a[2 * j]     = (_Float16)(float)c0;
                    a[2 * j + 1] = (_Float16)(float)c1;
                }
                acc = __builtin_amdgcn_wmma_f32_16x16x32_f16(
                        false, a, false, ones, (short)0, acc, false, false);
            }
            // D layout: lane 0 holds M=t*16+0..7 in VGPR 0..7 (N=0),
            //           lane 16 holds M=t*16+8..15. B=ones => all N identical.
            if (lane == 0) {
                #pragma unroll
                for (int r = 0; r < 8; ++r) sums[t * 16 + r] = acc[r];
            } else if (lane == 16) {
                #pragma unroll
                for (int r = 0; r < 8; ++r) sums[t * 16 + 8 + r] = acc[r];
            }
        }

        // Output permutation: col0 = bin0, col1 = bin29, col j = bin j-1 (j>=2)
        if (lane < NBINS) {
            int nat = (lane == 0) ? 0 : (lane == 1) ? (NBINS - 1) : (lane - 1);
            out[(size_t)row * NBINS + lane] = sums[nat];
        }
    }
}

extern "C" void kernel_launch(void* const* d_in, const int* in_sizes, int n_in,
                              void* d_out, int out_size, void* d_ws, size_t ws_size,
                              hipStream_t stream)
{
    const float* x   = (const float*)d_in[0];
    float*       out = (float*)d_out;

    int rows = out_size / NBINS;            // 4096
    if (rows <= 0) return;
    int cols = in_sizes[0] / rows;          // 16384

    hist_rows_kernel<<<dim3(rows), dim3(256), 0, stream>>>(x, out, cols);
}
